// DeepTimeGraphNet_64183991272030
// MI455X (gfx1250) — matrix operand order repaired and checked by reading, not compiled, not run on hardware
//
#include <hip/hip_runtime.h>

typedef __attribute__((ext_vector_type(16))) _Float16 v16h;
typedef __attribute__((ext_vector_type(8)))  _Float16 v8h;
typedef __attribute__((ext_vector_type(8)))  float    v8f;

constexpr int kBS   = 64;
constexpr int kN    = 1024;
constexpr int kT    = 1200;
constexpr int kRows = kBS * kN;   // 65536
constexpr int kRPB  = 8;          // rows (waves) per block

// ---------------------------------------------------------------------------
// Kernel 1: streaming conv/pool chain.  One wave32 per row of 1200 floats.
// conv0(k2,s2) + maxpool3 have combined stride 6 => m1[j] reads x[6j..6j+5]
// disjointly: 3x GLOBAL_LOAD_B64 per lane, fully coalesced across the wave.
// The rest of the pyramid (200->50->12->3->1) is staged through LDS.
// Bandwidth-bound: 314.6 MB of x at 23.3 TB/s ~= 13.5 us floor.
// ---------------------------------------------------------------------------
__global__ __launch_bounds__(256) void conv_chain_kernel(
    const float* __restrict__ x,
    const float* __restrict__ c0_w, const float* __restrict__ c0_b,
    const float* __restrict__ c2_w, const float* __restrict__ c2_b,
    const float* __restrict__ c4_w, const float* __restrict__ c4_b,
    const float* __restrict__ c6_w, const float* __restrict__ c6_b,
    const float* __restrict__ c8_w, const float* __restrict__ c8_b,
    float* __restrict__ feat)
{
  __shared__ float m1s[kRPB][208];   // L=200 after conv0+pool3+relu
  __shared__ float m3s[kRPB][52];    // L=50  after conv2+pool2+relu
  __shared__ float m5s[kRPB][13];    // L=12  after conv4+pool2+relu
  __shared__ float m7s[kRPB][3];     // L=3   after conv6+pool2+relu

  const int wave = threadIdx.x >> 5;
  const int lane = threadIdx.x & 31;
  const int row  = blockIdx.x * kRPB + wave;          // < 65536

  const float* __restrict__ xr = x + (size_t)row * kT;

  // ---- Phase A: m1[0..199], disjoint 6-float windows of x -----------------
  {
    const float w00 = c0_w[0], w01 = c0_w[1], b0 = c0_b[0];
    #pragma unroll
    for (int j0 = 0; j0 < 200; j0 += 32) {
      int j = j0 + lane;
      if (j < 200) {
        const float2* p = (const float2*)(xr + 6 * j);
        float2 a = p[0], b = p[1], c = p[2];
        float y0 = fmaf(w01, a.y, fmaf(w00, a.x, b0));
        float y1 = fmaf(w01, b.y, fmaf(w00, b.x, b0));
        float y2 = fmaf(w01, c.y, fmaf(w00, c.x, b0));
        m1s[wave][j] = fmaxf(fmaxf(fmaxf(y0, y1), y2), 0.0f);
      }
    }
  }
  __syncthreads();

  // ---- Phase B: conv2(k4,s2,p1) on 200 -> 100, pool2+relu -> m3[0..49] ----
  {
    const float a0 = c2_w[0], a1 = c2_w[1], a2 = c2_w[2], a3 = c2_w[3], bb = c2_b[0];
    auto g = [&](int i) -> float { return ((unsigned)i < 200u) ? m1s[wave][i] : 0.0f; };
    auto y = [&](int i) -> float {
      return fmaf(a3, g(2*i+2), fmaf(a2, g(2*i+1), fmaf(a1, g(2*i), fmaf(a0, g(2*i-1), bb))));
    };
    for (int j = lane; j < 50; j += 32)
      m3s[wave][j] = fmaxf(fmaxf(y(2*j), y(2*j+1)), 0.0f);
  }
  __syncthreads();

  // ---- Phase C: conv4(k4,s2,p1) on 50 -> 25, pool2+relu -> m5[0..11] ------
  if (lane < 12) {
    const float a0 = c4_w[0], a1 = c4_w[1], a2 = c4_w[2], a3 = c4_w[3], bb = c4_b[0];
    auto g = [&](int i) -> float { return ((unsigned)i < 50u) ? m3s[wave][i] : 0.0f; };
    auto y = [&](int i) -> float {
      return fmaf(a3, g(2*i+2), fmaf(a2, g(2*i+1), fmaf(a1, g(2*i), fmaf(a0, g(2*i-1), bb))));
    };
    m5s[wave][lane] = fmaxf(fmaxf(y(2*lane), y(2*lane+1)), 0.0f);
  }
  __syncthreads();

  // ---- Phase D: conv6(k4,s2,p1) on 12 -> 6, pool2+relu -> m7[0..2] --------
  if (lane < 3) {
    const float a0 = c6_w[0], a1 = c6_w[1], a2 = c6_w[2], a3 = c6_w[3], bb = c6_b[0];
    auto g = [&](int i) -> float { return ((unsigned)i < 12u) ? m5s[wave][i] : 0.0f; };
    auto y = [&](int i) -> float {
      return fmaf(a3, g(2*i+2), fmaf(a2, g(2*i+1), fmaf(a1, g(2*i), fmaf(a0, g(2*i-1), bb))));
    };
    m7s[wave][lane] = fmaxf(fmaxf(y(2*lane), y(2*lane+1)), 0.0f);
  }
  __syncthreads();

  // ---- Phase E: conv8(k3,s1,p0) on 3 -> 1 ---------------------------------
  if (lane == 0) {
    feat[row] = fmaf(c8_w[2], m7s[wave][2],
                fmaf(c8_w[1], m7s[wave][1],
                fmaf(c8_w[0], m7s[wave][0], c8_b[0])));
  }
}

// ---------------------------------------------------------------------------
// Kernel 2: classifier GEMM with WMMA + softmax.
// logits(64x3) = feat(64x1024) @ cls_w^T(1024x3) + b, softmax over 3.
// 4 waves; wave w computes the 16x16 C-tile for batch rows 16w..16w+15
// (N padded 3 -> 16) via v_wmma_f32_16x16x32_f16, K looped 1024/32 = 32x.
// B is staged ONCE into LDS as f16 (zero-padded to 16 rows) so the K-loop
// reads are unconditional contiguous ds_load_b128 (no divergent scalar
// gathers).  Row stride 1032 halfs = 516 dwords => lane n starts on bank
// 4n mod 64: conflict-free across the 16 columns.
// ---------------------------------------------------------------------------
constexpr int kBStride = 1032;    // halfs per padded B row

__global__ __launch_bounds__(128) void cls_wmma_softmax_kernel(
    const float* __restrict__ feat,   // (64, 1024)
    const float* __restrict__ cls_w,  // (3, 1024) row-major
    const float* __restrict__ cls_b,  // (3)
    float* __restrict__ out)          // (64, 3)
{
  __shared__ _Float16 Bs[16 * kBStride];   // ~33 KB
  __shared__ float logits[64][4];

  const int tid  = threadIdx.x;
  const int wave = tid >> 5;               // M-tile index 0..3
  const int lane = tid & 31;
  const int hi   = (lane >= 16) ? 1 : 0;
  const int l15  = lane & 15;

  // ---- Stage B into LDS: Bs[n][k] = (n < 3) ? f16(cls_w[n][k]) : 0 --------
  for (int i = tid; i < 16 * kBStride; i += 128)
    Bs[i] = (_Float16)0.0f;
  __syncthreads();
  for (int i = tid; i < 3 * kN; i += 128) {
    int n = i >> 10;          // / 1024
    int k = i & (kN - 1);     // % 1024
    Bs[n * kBStride + k] = (_Float16)cls_w[i];
  }
  __syncthreads();

  // A: lanes 0-15 hold M=lane, lanes 16-31 hold M=lane-16.
  const float* __restrict__ arow = feat + (size_t)(wave * 16 + l15) * kN;
  const _Float16* __restrict__ brow = Bs + l15 * kBStride;

  v8f acc = {0.f, 0.f, 0.f, 0.f, 0.f, 0.f, 0.f, 0.f};

  for (int k0 = 0; k0 < kN; k0 += 32) {
    v16h a, b;
    // A operand, contiguous k order chosen here (B matches): elem e -> K = e + (hi ? 16 : 0)
    // => one contiguous 16-float run: [k0+hi*16, +16)
    #pragma unroll
    for (int e = 0; e < 16; ++e) {
      int K = k0 + e + (hi ? 16 : 0);
      a[e] = (_Float16)arow[K];
    }
    // B operand, same contiguous k order as A: N = lane&15; elem e -> K = e + (hi ? 16 : 0)
    // => 16 contiguous halfs from LDS, 16B-aligned: two v8h (B128) loads.
    const _Float16* bp = brow + k0 + hi * 16;
    v8h blo = *(const v8h*)(bp);
    v8h bhi = *(const v8h*)(bp + 8);
    #pragma unroll
    for (int e = 0; e < 8; ++e) { b[e] = blo[e]; b[e + 8] = bhi[e]; }

    acc = __builtin_amdgcn_wmma_f32_16x16x32_f16(
        /*neg_a=*/false, a, /*neg_b=*/false, b,
        /*c_mod=*/(short)0, acc, /*reuse_a=*/false, /*reuse_b=*/false);
  }

  // D layout: VGPR r, lane -> M = r + hi*8, N = lane&15
  #pragma unroll
  for (int r = 0; r < 8; ++r) {
    int M = r + hi * 8;
    if (l15 < 3) logits[wave * 16 + M][l15] = acc[r] + cls_b[l15];
  }
  __syncthreads();

  if (tid < 64) {
    float a0 = logits[tid][0];
    float a1 = logits[tid][1];
    float a2 = logits[tid][2];
    float mx = fmaxf(a0, fmaxf(a1, a2));
    float e0 = __expf(a0 - mx), e1 = __expf(a1 - mx), e2 = __expf(a2 - mx);
    float inv = 1.0f / (e0 + e1 + e2);
    out[tid * 3 + 0] = e0 * inv;
    out[tid * 3 + 1] = e1 * inv;
    out[tid * 3 + 2] = e2 * inv;
  }
}

// ---------------------------------------------------------------------------
extern "C" void kernel_launch(void* const* d_in, const int* in_sizes, int n_in,
                              void* d_out, int out_size, void* d_ws, size_t ws_size,
                              hipStream_t stream) {
  (void)in_sizes; (void)n_in; (void)out_size; (void)ws_size;
  const float* x    = (const float*)d_in[0];
  const float* c0w  = (const float*)d_in[1];
  const float* c0b  = (const float*)d_in[2];
  const float* c2w  = (const float*)d_in[3];
  const float* c2b  = (const float*)d_in[4];
  const float* c4w  = (const float*)d_in[5];
  const float* c4b  = (const float*)d_in[6];
  const float* c6w  = (const float*)d_in[7];
  const float* c6b  = (const float*)d_in[8];
  const float* c8w  = (const float*)d_in[9];
  const float* c8b  = (const float*)d_in[10];
  // d_in[11..14]: dead GCN params (unused by the reference's live dataflow)
  const float* clsw = (const float*)d_in[15];
  const float* clsb = (const float*)d_in[16];

  float* feat = (float*)d_ws;   // 65536 floats = 256 KB scratch

  conv_chain_kernel<<<kRows / kRPB, 256, 0, stream>>>(
      x, c0w, c0b, c2w, c2b, c4w, c4b, c6w, c6b, c8w, c8b, feat);

  cls_wmma_softmax_kernel<<<1, 128, 0, stream>>>(feat, clsw, clsb, (float*)d_out);
}
